// GPTLanguageModel_74921409511539
// MI455X (gfx1250) — matrix operand order repaired and checked by reading, not compiled, not run on hardware
//
#include <hip/hip_runtime.h>

// ---------------- model constants ----------------
#define BB 64
#define TT 256
#define VV 65
#define EE 384
#define HH 6
#define DH 64
#define LL 6
#define FF 1536
#define BT (BB * TT)      // 16384
#define VPAD 128          // LM head N padded to GEMM tile

typedef __attribute__((ext_vector_type(16))) _Float16 v16h;
typedef __attribute__((ext_vector_type(8)))  _Float16 v8h;
typedef __attribute__((ext_vector_type(8)))  float    v8f;

// ---------------- WMMA helpers (CDNA5 layouts, CDNA5 ISA) ----------------
__device__ __forceinline__ v16h cat16(v8h lo, v8h hi) {
  return __builtin_shufflevector(lo, hi, 0,1,2,3,4,5,6,7,8,9,10,11,12,13,14,15);
}

// A fragment 16x32 f16, A row-major [M x K] (lda = K elements).
// lane m = lane&15, half = lane>>4: elems 0..7 -> K = k0+8*half .. +7,
// elems 8..15 -> K = k0+16+8*half .. +7   (two 16B loads)
__device__ __forceinline__ v16h load_a_frag(const _Float16* A, int lda,
                                            int m0, int k0, int lane) {
  int m = lane & 15, hf = lane >> 4;
  const _Float16* p = A + (size_t)(m0 + m) * lda + k0 + 8 * hf;
  v8h lo = *(const v8h*)p;
  v8h hi = *(const v8h*)(p + 16);
  return cat16(lo, hi);
}

// B fragment 32x16 f16 from TRANSPOSED operand Bt [N x K] row-major (ldb = K).
// lane n = lane&15, half = lane>>4: elems 0..7 -> K = k0+8*half .. +7,
// elems 8..15 -> K = k0+16+8*half .. +7 (same order as A)
__device__ __forceinline__ v16h load_b_frag(const _Float16* Bt, int ldb,
                                            int n0, int k0, int lane) {
  int n = lane & 15, hf = lane >> 4;
  const _Float16* p = Bt + (size_t)(n0 + n) * ldb + k0 + 8 * hf;
  v8h lo = *(const v8h*)p;
  v8h hi = *(const v8h*)(p + 16);
  return cat16(lo, hi);
}

__device__ __forceinline__ v8f wmma16(v16h a, v16h b, v8f c) {
  return __builtin_amdgcn_wmma_f32_16x16x32_f16(false, a, false, b,
                                                (short)0, c, false, false);
}

// 16-lane butterfly reductions via v_permlane16_b32 (VALU, no LDS round-trip).
// Selector nibbles give lane-in-row source index; patterns implement xor-k.
__device__ __forceinline__ float perml16(float v, unsigned s0, unsigned s1) {
  int u = __builtin_bit_cast(int, v);
  int r = __builtin_amdgcn_permlane16(u, u, (int)s0, (int)s1, false, false);
  return __builtin_bit_cast(float, r);
}
__device__ __forceinline__ float rowmax16(float v) {
  v = fmaxf(v, perml16(v, 0x67452301u, 0xEFCDAB89u));  // xor 1
  v = fmaxf(v, perml16(v, 0x54761032u, 0xDCFE98BAu));  // xor 2
  v = fmaxf(v, perml16(v, 0x32107654u, 0xBA98FEDCu));  // xor 4
  v = fmaxf(v, perml16(v, 0xFEDCBA98u, 0x76543210u));  // xor 8
  return v;
}
__device__ __forceinline__ float rowsum16(float v) {
  v += perml16(v, 0x67452301u, 0xEFCDAB89u);
  v += perml16(v, 0x54761032u, 0xDCFE98BAu);
  v += perml16(v, 0x32107654u, 0xBA98FEDCu);
  v += perml16(v, 0xFEDCBA98u, 0x76543210u);
  return v;
}

// ------- weight convert+transpose, LDS-tiled: f32 [b,K,N] -> f16 [b,Npad,K] --
// grid (K/32, Npad/32, batch), block (32,8). Coalesced read along N,
// coalesced write along K; +1 pad column avoids LDS bank conflicts.
__global__ void convertT_kernel(const float* __restrict__ src,
                                _Float16* __restrict__ dst,
                                int K, int N, int Npad) {
  __shared__ float tile[32][33];
  int b  = blockIdx.z;
  int k0 = blockIdx.x * 32;
  int n0 = blockIdx.y * 32;
  int tx = threadIdx.x, ty = threadIdx.y;
#pragma unroll
  for (int i = ty; i < 32; i += 8) {
    int k = k0 + i, n = n0 + tx;
    tile[i][tx] = (n < N) ? src[((size_t)b * K + k) * N + n] : 0.0f;
  }
  __syncthreads();
#pragma unroll
  for (int i = ty; i < 32; i += 8) {
    int n = n0 + i, k = k0 + tx;
    dst[((size_t)b * Npad + n) * K + k] = (_Float16)tile[tx][i];
  }
}

// ---------------- embedding: x = tok_emb[idx] + pos_emb ----------------
__global__ void embed_kernel(const int* __restrict__ idx,
                             const float* __restrict__ tok_emb,
                             const float* __restrict__ pos_emb,
                             float* __restrict__ x) {
  int i = blockIdx.x * 256 + threadIdx.x;   // over BT*E
  if (i >= BT * EE) return;
  int e  = i % EE;
  int bt = i / EE;
  int t  = bt % TT;
  x[i] = tok_emb[(size_t)idx[bt] * EE + e] + pos_emb[(size_t)t * EE + e];
}

// ---------------- layernorm: f32 in -> f16 out, one wave32 per row ---------
__global__ void ln_kernel(const float* __restrict__ x,
                          const float* __restrict__ g,
                          const float* __restrict__ bta,
                          _Float16* __restrict__ out) {
  int wave = threadIdx.x >> 5, lane = threadIdx.x & 31;
  int row  = blockIdx.x * 8 + wave;
  const float* xr = x + (size_t)row * EE;
  float vals[12];
  float s = 0.f;
#pragma unroll
  for (int i = 0; i < 12; ++i) { vals[i] = xr[lane + i * 32]; s += vals[i]; }
#pragma unroll
  for (int off = 16; off > 0; off >>= 1) s += __shfl_xor(s, off, 32);
  float mu = s * (1.0f / EE);
  float vs = 0.f;
#pragma unroll
  for (int i = 0; i < 12; ++i) { float d = vals[i] - mu; vs += d * d; }
#pragma unroll
  for (int off = 16; off > 0; off >>= 1) vs += __shfl_xor(vs, off, 32);
  float rstd = rsqrtf(vs * (1.0f / EE) + 1e-5f);
  _Float16* orow = out + (size_t)row * EE;
#pragma unroll
  for (int i = 0; i < 12; ++i) {
    int c = lane + i * 32;
    orow[c] = (_Float16)((vals[i] - mu) * rstd * g[c] + bta[c]);
  }
}

// ---------------- generic WMMA GEMM, software-pipelined K loop ----------------
// C[M,N] = A[M,K](f16,row-major) @ Bt[Npad,K]^T (+bias)
// Block 256 threads = 8 waves; wave tile 32(M)x64(N); block tile 128x128.
// Fragments for step k+32 are prefetched before the 8 WMMAs of step k, so the
// loads have a full WMMA group of latency to complete (no per-step stall).
// MODE 0: f16 out; 1: f16 relu out; 2: f32 out = resid + v; 3: f32 out, col<Nvalid
template <int MODE>
__global__ void gemm_kernel(const _Float16* __restrict__ A,
                            const _Float16* __restrict__ Bt,
                            const float* __restrict__ bias,
                            void* __restrict__ Cout,
                            const float* __restrict__ resid,
                            int K, int ldc, int Nvalid) {
  int tid = threadIdx.x, lane = tid & 31, w = tid >> 5;
  int wm = w & 3, wn = w >> 2;
  int m0 = blockIdx.x * 128 + wm * 32;
  int n0 = blockIdx.y * 128 + wn * 64;
  v8f acc[2][4] = {};
  v16h a0 = load_a_frag(A, K, m0,      0, lane);
  v16h a1 = load_a_frag(A, K, m0 + 16, 0, lane);
  v16h b0 = load_b_frag(Bt, K, n0,      0, lane);
  v16h b1 = load_b_frag(Bt, K, n0 + 16, 0, lane);
  v16h b2 = load_b_frag(Bt, K, n0 + 32, 0, lane);
  v16h b3 = load_b_frag(Bt, K, n0 + 48, 0, lane);
  for (int k0 = 0; k0 < K; k0 += 32) {
    int kn = (k0 + 32 < K) ? k0 + 32 : 0;   // last prefetch wraps (harmless)
    v16h na0 = load_a_frag(A, K, m0,      kn, lane);
    v16h na1 = load_a_frag(A, K, m0 + 16, kn, lane);
    v16h nb0 = load_b_frag(Bt, K, n0,      kn, lane);
    v16h nb1 = load_b_frag(Bt, K, n0 + 16, kn, lane);
    v16h nb2 = load_b_frag(Bt, K, n0 + 32, kn, lane);
    v16h nb3 = load_b_frag(Bt, K, n0 + 48, kn, lane);
    acc[0][0] = wmma16(a0, b0, acc[0][0]);
    acc[1][0] = wmma16(a1, b0, acc[1][0]);
    acc[0][1] = wmma16(a0, b1, acc[0][1]);
    acc[1][1] = wmma16(a1, b1, acc[1][1]);
    acc[0][2] = wmma16(a0, b2, acc[0][2]);
    acc[1][2] = wmma16(a1, b2, acc[1][2]);
    acc[0][3] = wmma16(a0, b3, acc[0][3]);
    acc[1][3] = wmma16(a1, b3, acc[1][3]);
    a0 = na0; a1 = na1; b0 = nb0; b1 = nb1; b2 = nb2; b3 = nb3;
  }
  int hf = lane >> 4, ln = lane & 15;
#pragma unroll
  for (int mt = 0; mt < 2; ++mt) {
#pragma unroll
    for (int nt = 0; nt < 4; ++nt) {
#pragma unroll
      for (int r = 0; r < 8; ++r) {
        int row = m0 + mt * 16 + r + 8 * hf;
        int col = n0 + nt * 16 + ln;
        float v = acc[mt][nt][r] + (bias ? bias[col] : 0.0f);
        size_t off = (size_t)row * ldc + col;
        if (MODE == 0) ((_Float16*)Cout)[off] = (_Float16)v;
        else if (MODE == 1) ((_Float16*)Cout)[off] = (_Float16)fmaxf(v, 0.0f);
        else if (MODE == 2) ((float*)Cout)[off] = resid[off] + v;
        else if (col < Nvalid) ((float*)Cout)[off] = v;
      }
    }
  }
}

// ------- transpose V, LDS-tiled: [BT,E] f16 -> [B,H,DH,T] f16 ----------------
// grid (T/32, DH/32, B*H), block (32,8). Both sides coalesced.
__global__ void vtrans_kernel(const _Float16* __restrict__ v,
                              _Float16* __restrict__ vT) {
  __shared__ _Float16 tile[32][33];
  int bh = blockIdx.z;
  int b = bh / HH, h = bh % HH;
  int t0 = blockIdx.x * 32, d0 = blockIdx.y * 32;
  int tx = threadIdx.x, ty = threadIdx.y;
  const _Float16* src = v + (size_t)b * TT * EE + h * DH;
#pragma unroll
  for (int i = ty; i < 32; i += 8)
    tile[i][tx] = src[(size_t)(t0 + i) * EE + d0 + tx];
  __syncthreads();
  _Float16* dst = vT + (size_t)bh * DH * TT;
#pragma unroll
  for (int i = ty; i < 32; i += 8)
    dst[(size_t)(d0 + i) * TT + t0 + tx] = tile[tx][i];
}

// ---------------- flash attention, one block per (b,head) ----------------
__global__ void attn_kernel(const _Float16* __restrict__ q16,
                            const _Float16* __restrict__ k16,
                            const _Float16* __restrict__ vT16,
                            _Float16* __restrict__ att16) {
  int bh = blockIdx.x;
  int b = bh / HH, h = bh % HH;
  const _Float16* q  = q16 + (size_t)b * TT * EE + h * DH;
  const _Float16* k  = k16 + (size_t)b * TT * EE + h * DH;
  const _Float16* vT = vT16 + (size_t)bh * DH * TT;
  _Float16* out = att16 + (size_t)b * TT * EE + h * DH;

  int tid = threadIdx.x, wave = tid >> 5, lane = tid & 31;
  int hf = lane >> 4, ln = lane & 15;
  __shared__ __align__(16) _Float16 plds[8][16 * 32];
  _Float16* myp = plds[wave];
  const float scale = 0.125f;  // 1/sqrt(64)

  for (int qt = 0; qt < 2; ++qt) {
    int qr = wave * 32 + qt * 16;       // q tile row base
    v16h aq0 = load_a_frag(q, EE, qr, 0, lane);
    v16h aq1 = load_a_frag(q, EE, qr, 32, lane);
    v8f acc0 = {}, acc1 = {}, acc2 = {}, acc3 = {};
    float mrun[8], lrun[8];
#pragma unroll
    for (int r = 0; r < 8; ++r) { mrun[r] = -1e30f; lrun[r] = 0.f; }

    for (int j0 = 0; j0 <= qr + 15; j0 += 32) {
      // scores for keys [j0, j0+32)
      v8f sA = {}, sB = {};
      sA = wmma16(aq0, load_b_frag(k, EE, j0, 0, lane), sA);
      sA = wmma16(aq1, load_b_frag(k, EE, j0, 32, lane), sA);
      sB = wmma16(aq0, load_b_frag(k, EE, j0 + 16, 0, lane), sB);
      sB = wmma16(aq1, load_b_frag(k, EE, j0 + 16, 32, lane), sB);

#pragma unroll
      for (int r = 0; r < 8; ++r) {
        int row = qr + r + 8 * hf;
        float vA = (j0 + ln      <= row) ? sA[r] * scale : -1e30f;
        float vB = (j0 + 16 + ln <= row) ? sB[r] * scale : -1e30f;
        float mt = rowmax16(fmaxf(vA, vB));       // VALU-only 16-lane butterfly
        float mnew = fmaxf(mrun[r], mt);
        float pA = __expf(vA - mnew);
        float pB = __expf(vB - mnew);
        float lsum = rowsum16(pA + pB);           // VALU-only 16-lane butterfly
        float sf = __expf(mrun[r] - mnew);
        mrun[r] = mnew;
        lrun[r] = lrun[r] * sf + lsum;
        acc0[r] *= sf; acc1[r] *= sf; acc2[r] *= sf; acc3[r] *= sf;
        // write P tile (C-layout -> LDS) for A-frag re-read
        myp[(r + 8 * hf) * 32 + ln]      = (_Float16)pA;
        myp[(r + 8 * hf) * 32 + 16 + ln] = (_Float16)pB;
      }
      // re-read P as A fragment (lane = row, halves = K groups); same-wave
      // LDS RAW is in-order per DScnt rules.
      v8h plo = *(const v8h*)&myp[ln * 32 + 8 * hf];
      v8h phi = *(const v8h*)&myp[ln * 32 + 16 + 8 * hf];
      v16h pf = cat16(plo, phi);
      acc0 = wmma16(pf, load_b_frag(vT, TT, 0,  j0, lane), acc0);
      acc1 = wmma16(pf, load_b_frag(vT, TT, 16, j0, lane), acc1);
      acc2 = wmma16(pf, load_b_frag(vT, TT, 32, j0, lane), acc2);
      acc3 = wmma16(pf, load_b_frag(vT, TT, 48, j0, lane), acc3);
    }

#pragma unroll
    for (int r = 0; r < 8; ++r) {
      int row = qr + r + 8 * hf;
      float inv = 1.0f / lrun[r];
      _Float16* orow = out + (size_t)row * EE;
      orow[ln]      = (_Float16)(acc0[r] * inv);
      orow[16 + ln] = (_Float16)(acc1[r] * inv);
      orow[32 + ln] = (_Float16)(acc2[r] * inv);
      orow[48 + ln] = (_Float16)(acc3[r] * inv);
    }
  }
}

// ---------------- loss: deterministic two-stage reduction ----------------
__global__ void loss_partial_kernel(const float* __restrict__ logits,
                                    const int* __restrict__ targets,
                                    float* __restrict__ partial) {
  int row = blockIdx.x * 256 + threadIdx.x;
  const float* lr = logits + (size_t)row * VV;
  float m = -1e30f;
  for (int j = 0; j < VV; ++j) m = fmaxf(m, lr[j]);
  float s = 0.f;
  for (int j = 0; j < VV; ++j) s += __expf(lr[j] - m);
  float nll = -(lr[targets[row]] - m - __logf(s));
  __shared__ float red[256];
  red[threadIdx.x] = nll;
  __syncthreads();
  for (int st = 128; st > 0; st >>= 1) {
    if (threadIdx.x < st) red[threadIdx.x] += red[threadIdx.x + st];
    __syncthreads();
  }
  if (threadIdx.x == 0) partial[blockIdx.x] = red[0];
}

__global__ void loss_final_kernel(const float* __restrict__ partial,
                                  float* __restrict__ loss) {
  __shared__ float red[64];
  red[threadIdx.x] = partial[threadIdx.x];
  __syncthreads();
  for (int st = 32; st > 0; st >>= 1) {
    if (threadIdx.x < st) red[threadIdx.x] += red[threadIdx.x + st];
    __syncthreads();
  }
  if (threadIdx.x == 0) *loss = red[0] * (1.0f / BT);
}

// ---------------- host orchestration ----------------
static inline size_t ws_align(size_t x) { return (x + 255) & ~(size_t)255; }

extern "C" void kernel_launch(void* const* d_in, const int* in_sizes, int n_in,
                              void* d_out, int out_size, void* d_ws, size_t ws_size,
                              hipStream_t stream) {
  (void)in_sizes; (void)n_in; (void)out_size; (void)ws_size;
  const int*   idx     = (const int*)d_in[0];
  const int*   targets = (const int*)d_in[1];
  const float* tok_emb = (const float*)d_in[2];
  const float* pos_emb = (const float*)d_in[3];
  const float* Wq  = (const float*)d_in[4];
  const float* Wk  = (const float*)d_in[5];
  const float* Wv  = (const float*)d_in[6];
  const float* Wo  = (const float*)d_in[7];
  const float* bo  = (const float*)d_in[8];
  const float* W1  = (const float*)d_in[9];
  const float* b1  = (const float*)d_in[10];
  const float* W2  = (const float*)d_in[11];
  const float* b2  = (const float*)d_in[12];
  const float* ln1_g = (const float*)d_in[13];
  const float* ln1_b = (const float*)d_in[14];
  const float* ln2_g = (const float*)d_in[15];
  const float* ln2_b = (const float*)d_in[16];
  const float* lnf_g = (const float*)d_in[17];
  const float* lnf_b = (const float*)d_in[18];
  const float* Wlm = (const float*)d_in[19];
  const float* blm = (const float*)d_in[20];

  float* logits = (float*)d_out;                 // [BT, VV]
  float* loss   = (float*)d_out + (size_t)BT * VV;

  // workspace layout
  char* ws = (char*)d_ws;
  size_t off = 0;
  auto take = [&](size_t bytes) { char* p = ws + off; off += ws_align(bytes); return p; };
  float*    x     = (float*)   take((size_t)BT * EE * 4);
  _Float16* h16   = (_Float16*)take((size_t)BT * EE * 2);
  _Float16* q16   = (_Float16*)take((size_t)BT * EE * 2);
  _Float16* k16   = (_Float16*)take((size_t)BT * EE * 2);
  _Float16* v16   = (_Float16*)take((size_t)BT * EE * 2);
  _Float16* vT16  = (_Float16*)take((size_t)BT * EE * 2);
  _Float16* att16 = (_Float16*)take((size_t)BT * EE * 2);
  _Float16* ffn16 = (_Float16*)take((size_t)BT * FF * 2);
  _Float16* WqT = (_Float16*)take((size_t)LL * EE * EE * 2);
  _Float16* WkT = (_Float16*)take((size_t)LL * EE * EE * 2);
  _Float16* WvT = (_Float16*)take((size_t)LL * EE * EE * 2);
  _Float16* WoT = (_Float16*)take((size_t)LL * EE * EE * 2);
  _Float16* W1T = (_Float16*)take((size_t)LL * EE * FF * 2);
  _Float16* W2T = (_Float16*)take((size_t)LL * FF * EE * 2);
  _Float16* WlmT = (_Float16*)take((size_t)VPAD * EE * 2);
  float* partial = (float*)take(64 * 4);

  auto cvt = [&](const float* src, _Float16* dst, int K, int N, int Npad, int batch) {
    convertT_kernel<<<dim3(K / 32, Npad / 32, batch), dim3(32, 8), 0, stream>>>(
        src, dst, K, N, Npad);
  };
  // weights -> f16 transposed
  cvt(Wq,  WqT,  EE, EE, EE,   LL);
  cvt(Wk,  WkT,  EE, EE, EE,   LL);
  cvt(Wv,  WvT,  EE, EE, EE,   LL);
  cvt(Wo,  WoT,  EE, EE, EE,   LL);
  cvt(W1,  W1T,  EE, FF, FF,   LL);
  cvt(W2,  W2T,  FF, EE, EE,   LL);
  cvt(Wlm, WlmT, EE, VV, VPAD, 1);

  // embedding
  embed_kernel<<<dim3((BT * EE + 255) / 256), dim3(256), 0, stream>>>(
      idx, tok_emb, pos_emb, x);

  dim3 blk(256);
  dim3 gE(BT / 128, EE / 128);    // N = 384
  dim3 gF(BT / 128, FF / 128);    // N = 1536
  dim3 gV(BT / 128, VPAD / 128);  // N = 128 (padded from 65)
  dim3 gLN(BT / 8);
  dim3 gVT(TT / 32, DH / 32, BB * HH);

  for (int l = 0; l < LL; ++l) {
    // ln1 -> h16
    ln_kernel<<<gLN, blk, 0, stream>>>(x, ln1_g + l * EE, ln1_b + l * EE, h16);
    // q,k,v projections (no bias)
    gemm_kernel<0><<<gE, blk, 0, stream>>>(h16, WqT + (size_t)l * EE * EE,
                                           nullptr, q16, nullptr, EE, EE, EE);
    gemm_kernel<0><<<gE, blk, 0, stream>>>(h16, WkT + (size_t)l * EE * EE,
                                           nullptr, k16, nullptr, EE, EE, EE);
    gemm_kernel<0><<<gE, blk, 0, stream>>>(h16, WvT + (size_t)l * EE * EE,
                                           nullptr, v16, nullptr, EE, EE, EE);
    // V transpose for P@V B-operand layout
    vtrans_kernel<<<gVT, dim3(32, 8), 0, stream>>>(v16, vT16);
    // flash attention
    attn_kernel<<<dim3(BB * HH), blk, 0, stream>>>(q16, k16, vT16, att16);
    // output projection + residual
    gemm_kernel<2><<<gE, blk, 0, stream>>>(att16, WoT + (size_t)l * EE * EE,
                                           bo + l * EE, x, x, EE, EE, EE);
    // ln2 -> h16
    ln_kernel<<<gLN, blk, 0, stream>>>(x, ln2_g + l * EE, ln2_b + l * EE, h16);
    // ffn1 (relu, f16 out)
    gemm_kernel<1><<<gF, blk, 0, stream>>>(h16, W1T + (size_t)l * EE * FF,
                                           b1 + l * FF, ffn16, nullptr, EE, FF, FF);
    // ffn2 + residual
    gemm_kernel<2><<<gE, blk, 0, stream>>>(ffn16, W2T + (size_t)l * FF * EE,
                                           b2 + l * EE, x, x, FF, EE, EE);
  }

  // final LN + LM head into d_out logits (bounds-checked at V=65)
  ln_kernel<<<gLN, blk, 0, stream>>>(x, lnf_g, lnf_b, h16);
  gemm_kernel<3><<<gV, blk, 0, stream>>>(h16, WlmT, blm, logits, nullptr,
                                         EE, VV, VV);

  // loss
  loss_partial_kernel<<<dim3(BT / 256), blk, 0, stream>>>(logits, targets, partial);
  loss_final_kernel<<<dim3(1), dim3(64), 0, stream>>>(partial, loss);
}
